// Attention_23759759081743
// MI455X (gfx1250) — matrix operand order repaired and checked by reading, not compiled, not run on hardware
//
#include <hip/hip_runtime.h>
#include <hip/hip_bf16.h>
#include <stdint.h>
#include <stddef.h>

// ---------------------------------------------------------------------------
// CDNA5 (gfx1250) attention block, all matmuls on v_wmma_f32_16x16x32_bf16.
//   kv = x@W_attn+b -> k [b,h,s,hd] bf16, vT [b,h,hd,s] bf16
//   q  = (y@W_mlp+b)*0.125 -> [b,h,64,hd] bf16 (rows 49..63 zero)
//   e  = exp(q k^T)*mask ; a = (e@v)/(sum e + eps) -> bf16
//   out = a@W_proj + b_proj -> f32
// Fragments use k = 16h+s: A as two 16B reads of permuted-K LDS tiles,
// B via transposed weights / k rows / vT rows in global memory.
// kv GEMM: 128x128 block tile, 8 waves, N on blockIdx.x for L2 reuse of x.
// ---------------------------------------------------------------------------

typedef __attribute__((ext_vector_type(16))) __bf16 v16bf;
typedef __attribute__((ext_vector_type(8)))  __bf16 v8bf;
typedef __attribute__((ext_vector_type(8)))  float  v8f;

union bf16x16 { v16bf v; v8bf h[2]; };

#define NHEAD   16
#define HD      64
#define SLEN    4096
#define DMODEL  1024
#define QLEN    49
#define QPAD    64
#define BATCH   16

#define WMMA_BF16(A, B, C) \
  __builtin_amdgcn_wmma_f32_16x16x32_bf16(false, (A), false, (B), (short)0, (C), false, false)

// Native RNE f32->bf16 (lowers to hardware cvt on gfx1250).
__device__ __forceinline__ __bf16 to_bf16(float f) { return (__bf16)f; }

// K-permutation for A-fragment-order storage: swap bits 3 and 4 of k&31.
__device__ __forceinline__ int kperm(int c) {
  return (c & ~24) | ((c & 8) << 1) | ((c & 16) >> 1);
}

__device__ __forceinline__ v8f v8f_zero() {
  v8f z;
#pragma unroll
  for (int i = 0; i < 8; ++i) z[i] = 0.0f;
  return z;
}

// ---------------------------------------------------------------------------
// Tiled transpose + f32->bf16: dst[c][r] = bf16(src[r][c]).  rows,cols % 32 == 0.
// ---------------------------------------------------------------------------
__global__ __launch_bounds__(256)
void transpose_cvt_kernel(const float* __restrict__ src, __bf16* __restrict__ dst,
                          int rows, int cols) {
  __shared__ __bf16 tile[32][33];
  const int c0 = blockIdx.x * 32, r0 = blockIdx.y * 32;
  const int tx = threadIdx.x, ty = threadIdx.y;        // 32 x 8
#pragma unroll
  for (int j = 0; j < 32; j += 8)
    tile[ty + j][tx] = to_bf16(src[(size_t)(r0 + ty + j) * cols + c0 + tx]);
  __syncthreads();
#pragma unroll
  for (int j = 0; j < 32; j += 8)
    dst[(size_t)(c0 + ty + j) * rows + r0 + tx] = tile[tx][ty + j];
}

// ---------------------------------------------------------------------------
// kv GEMM: (B*S x D) @ (D x 2D) + b_attn -> k [b,h,s,hd], vT [b,h,hd,s].
// 256 thr (8 waves), tile 128x128, wave grid 4M x 2N, K-step 32.
// blockIdx.x = N (fast) so x strips stay resident in L2.
// ---------------------------------------------------------------------------
__global__ __launch_bounds__(256)
void gemm_kv_kernel(const float* __restrict__ x, const __bf16* __restrict__ WT,
                    const float* __restrict__ b_attn,
                    __bf16* __restrict__ kout, __bf16* __restrict__ vT) {
  __shared__ __bf16 As[128][40];                        // permuted-K A tile
  const int tid = threadIdx.x;
  const int lane = tid & 31, w = tid >> 5;
  const int hi = lane >> 4, ln = lane & 15;
  const int wr = w & 3, wc = w >> 2;                    // wave row/col
  const int n0 = blockIdx.x * 128;
  const int m0 = blockIdx.y * 128;
  const int ar = tid >> 1;                              // staging row 0..127
  const int ac = (tid & 1) * 16;                        // staging col base
  const int p0 = (ac >> 1) & 8;                         // permuted store base

  v8f acc[2][4];
#pragma unroll
  for (int mi = 0; mi < 2; ++mi)
#pragma unroll
    for (int n = 0; n < 4; ++n) acc[mi][n] = v8f_zero();

  for (int k0 = 0; k0 < DMODEL; k0 += 32) {
    const float* xp = &x[(size_t)(m0 + ar) * DMODEL + k0 + ac];
    v8bf lo, hv;
#pragma unroll
    for (int e = 0; e < 8; ++e) lo[e] = to_bf16(xp[e]);
#pragma unroll
    for (int e = 0; e < 8; ++e) hv[e] = to_bf16(xp[8 + e]);
    *(v8bf*)&As[ar][p0]      = lo;
    *(v8bf*)&As[ar][p0 + 16] = hv;
    if (k0 + 32 < DMODEL)
      __builtin_prefetch(xp + 32, 0, 1);                // global_prefetch_b8
    __syncthreads();

    bf16x16 au[2];
#pragma unroll
    for (int mi = 0; mi < 2; ++mi) {
      const int row = wr * 32 + mi * 16 + ln;
      au[mi].h[0] = *(const v8bf*)&As[row][hi * 8];
      au[mi].h[1] = *(const v8bf*)&As[row][hi * 8 + 16];
    }
#pragma unroll
    for (int n = 0; n < 4; ++n) {
      const v16bf bfg = *(const v16bf*)
          &WT[(size_t)(n0 + wc * 64 + n * 16 + ln) * DMODEL + k0 + hi * 16];
#pragma unroll
      for (int mi = 0; mi < 2; ++mi)
        acc[mi][n] = WMMA_BF16(au[mi].v, bfg, acc[mi][n]);
    }
    __syncthreads();
  }

#pragma unroll
  for (int n = 0; n < 4; ++n) {
    int gn = n0 + wc * 64 + n * 16 + ln;
    float bias = b_attn[gn];
#pragma unroll
    for (int mi = 0; mi < 2; ++mi) {
#pragma unroll
      for (int i = 0; i < 8; ++i) {
        int gm = m0 + wr * 32 + mi * 16 + i + hi * 8;   // row in B*S
        int b = gm >> 12, s = gm & (SLEN - 1);
        float val = acc[mi][n][i] + bias;
        if (gn < DMODEL) {
          int h = gn >> 6, d = gn & 63;
          kout[((size_t)(b * NHEAD + h) * SLEN + s) * HD + d] = to_bf16(val);
        } else {
          int gn2 = gn - DMODEL;
          int h = gn2 >> 6, d = gn2 & 63;
          vT[((size_t)(b * NHEAD + h) * HD + d) * SLEN + s] = to_bf16(val);
        }
      }
    }
  }
}

// ---------------------------------------------------------------------------
// q GEMM: padded (B*64 x D) @ (D x D) + b_mlp, *0.125 -> q [b,h,64,hd] bf16.
// 128 thr (4 waves), tile 64x64.
// ---------------------------------------------------------------------------
__global__ __launch_bounds__(128)
void gemm_q_kernel(const float* __restrict__ y, const __bf16* __restrict__ WT,
                   const float* __restrict__ b_mlp, __bf16* __restrict__ qout) {
  __shared__ __bf16 As[64][40];
  const int tid = threadIdx.x;
  const int lane = tid & 31, w = tid >> 5;
  const int hi = lane >> 4, ln = lane & 15;
  const int n0 = blockIdx.x * 64;
  const int m0 = blockIdx.y * 64;
  const int ar = tid >> 1, ac = (tid & 1) * 16;
  const int p0 = (ac >> 1) & 8;
  const int pr = m0 + ar, pb = pr >> 6, pqi = pr & 63;

  v8f acc[4];
#pragma unroll
  for (int n = 0; n < 4; ++n) acc[n] = v8f_zero();

  for (int k0 = 0; k0 < DMODEL; k0 += 32) {
    v8bf lo, hv;
    if (pqi < QLEN) {
      const float* yp = &y[(size_t)(pb * QLEN + pqi) * DMODEL + k0 + ac];
#pragma unroll
      for (int e = 0; e < 8; ++e) lo[e] = to_bf16(yp[e]);
#pragma unroll
      for (int e = 0; e < 8; ++e) hv[e] = to_bf16(yp[8 + e]);
    } else {
#pragma unroll
      for (int e = 0; e < 8; ++e) { lo[e] = (__bf16)0.0f; hv[e] = (__bf16)0.0f; }
    }
    *(v8bf*)&As[ar][p0]      = lo;
    *(v8bf*)&As[ar][p0 + 16] = hv;
    __syncthreads();

    bf16x16 au;
    au.h[0] = *(const v8bf*)&As[w * 16 + ln][hi * 8];
    au.h[1] = *(const v8bf*)&As[w * 16 + ln][hi * 8 + 16];
#pragma unroll
    for (int n = 0; n < 4; ++n) {
      const v16bf bfg =
          *(const v16bf*)&WT[(size_t)(n0 + n * 16 + ln) * DMODEL + k0 + hi * 16];
      acc[n] = WMMA_BF16(au.v, bfg, acc[n]);
    }
    __syncthreads();
  }

#pragma unroll
  for (int n = 0; n < 4; ++n) {
    int gn = n0 + n * 16 + ln;
    float bias = b_mlp[gn];
    int h = gn >> 6, d = gn & 63;
#pragma unroll
    for (int i = 0; i < 8; ++i) {
      int gm = m0 + w * 16 + i + hi * 8;
      int b = gm >> 6, qi = gm & 63;
      float val = (qi < QLEN) ? (acc[n][i] + bias) * 0.125f : 0.0f;
      qout[((size_t)(b * NHEAD + h) * QPAD + qi) * HD + d] = to_bf16(val);
    }
  }
}

// ---------------------------------------------------------------------------
// Attention: one block per (b,h), 256 thr (8 waves).
// Scores B-frag: contiguous k rows. PV B-frag: contiguous vT rows.
// Qs/Es held in LDS in permuted-K (fragment) order.
// ---------------------------------------------------------------------------
__global__ __launch_bounds__(256)
void attention_kernel(const __bf16* __restrict__ kin,
                      const __bf16* __restrict__ vT,
                      const __bf16* __restrict__ qin,
                      const float* __restrict__ mask,
                      __bf16* __restrict__ aout) {
  __shared__ __bf16 Qs[QPAD][72];
  __shared__ __bf16 Es[QPAD][72];
  __shared__ float  den[QPAD];

  const int tid = threadIdx.x;
  const int lane = tid & 31, w = tid >> 5;
  const int hi = lane >> 4, ln = lane & 15;
  const int bh = blockIdx.x;                            // b*16 + h
  const int b = bh >> 4, h = bh & 15;
  const __bf16* kb  = kin + (size_t)bh * SLEN * HD;
  const __bf16* vbT = vT  + (size_t)bh * HD * SLEN;
  const __bf16* qb  = qin + (size_t)bh * QPAD * HD;

  {                                                     // stage q (permuted)
    int r = tid >> 2, c0 = (tid & 3) * 16;
    bf16x16 u;
    u.v = *(const v16bf*)&qb[(size_t)r * HD + c0];
    int base = (c0 & 32) | ((c0 & 16) ? 8 : 0);
    *(v8bf*)&Qs[r][base]      = u.h[0];
    *(v8bf*)&Qs[r][base + 16] = u.h[1];
  }
  if (tid < QPAD) den[tid] = 0.0f;
  __syncthreads();

  v8f acc[2];
  acc[0] = v8f_zero();
  acc[1] = v8f_zero();

  for (int s0 = 0; s0 < SLEN; s0 += 64) {
#pragma unroll
    for (int t = 0; t < 2; ++t) {                       // scores: 2 tiles/wave
      int tt = w * 2 + t;
      int mi = tt >> 2, nj = tt & 3;
      v8f sc = v8f_zero();
#pragma unroll
      for (int k0 = 0; k0 < HD; k0 += 32) {
        bf16x16 au;
        au.h[0] = *(const v8bf*)&Qs[mi * 16 + ln][k0 + hi * 8];
        au.h[1] = *(const v8bf*)&Qs[mi * 16 + ln][k0 + hi * 8 + 16];
        const v16bf bfg =
            *(const v16bf*)&kb[(size_t)(s0 + nj * 16 + ln) * HD + k0 + hi * 16];
        sc = WMMA_BF16(au.v, bfg, sc);
      }
      int col = nj * 16 + ln;                           // exp*mask -> Es (permuted)
      int pcol = (col & 32) | kperm(col & 31);
      int scol = s0 + col;
#pragma unroll
      for (int i = 0; i < 8; ++i) {
        int r = mi * 16 + i + hi * 8;
        float ev = 0.0f;
        if (r < QLEN) ev = __expf(sc[i]) * mask[(size_t)r * SLEN + scol];
        Es[r][pcol] = to_bf16(ev);
        atomicAdd(&den[r], ev);
      }
    }
    __syncthreads();

#pragma unroll
    for (int t = 0; t < 2; ++t) {                       // PV: num += e @ v
      int tt = w * 2 + t;
      int mi = tt >> 2, ni = tt & 3;
#pragma unroll
      for (int k0 = 0; k0 < 64; k0 += 32) {
        bf16x16 au;
        au.h[0] = *(const v8bf*)&Es[mi * 16 + ln][k0 + hi * 8];
        au.h[1] = *(const v8bf*)&Es[mi * 16 + ln][k0 + hi * 8 + 16];
        const v16bf bfg =
            *(const v16bf*)&vbT[(size_t)(ni * 16 + ln) * SLEN + s0 + k0 + hi * 16];
        acc[t] = WMMA_BF16(au.v, bfg, acc[t]);
      }
    }
    __syncthreads();
  }

#pragma unroll
  for (int t = 0; t < 2; ++t) {
    int tt = w * 2 + t;
    int mi = tt >> 2, ni = tt & 3;
#pragma unroll
    for (int i = 0; i < 8; ++i) {
      int r = mi * 16 + i + hi * 8;
      float val = acc[t][i] / (den[r] + 1e-6f);
      aout[(size_t)(b * QPAD + r) * DMODEL + h * HD + ni * 16 + ln] = to_bf16(val);
    }
  }
}

// ---------------------------------------------------------------------------
// Output projection: padded (B*64 x D) @ (D x Dp) + b_proj -> f32 (qi<49).
// ---------------------------------------------------------------------------
__global__ __launch_bounds__(128)
void gemm_proj_kernel(const __bf16* __restrict__ a, const __bf16* __restrict__ WT,
                      const float* __restrict__ b_proj, float* __restrict__ out) {
  __shared__ __bf16 As[64][40];
  const int tid = threadIdx.x;
  const int lane = tid & 31, w = tid >> 5;
  const int hi = lane >> 4, ln = lane & 15;
  const int n0 = blockIdx.x * 64;
  const int m0 = blockIdx.y * 64;
  const int ar = tid >> 1, ac = (tid & 1) * 16;
  const int p0 = (ac >> 1) & 8;

  v8f acc[4];
#pragma unroll
  for (int n = 0; n < 4; ++n) acc[n] = v8f_zero();

  for (int k0 = 0; k0 < DMODEL; k0 += 32) {
    bf16x16 u;
    u.v = *(const v16bf*)&a[(size_t)(m0 + ar) * DMODEL + k0 + ac];
    *(v8bf*)&As[ar][p0]      = u.h[0];
    *(v8bf*)&As[ar][p0 + 16] = u.h[1];
    if (k0 + 32 < DMODEL)
      __builtin_prefetch(&a[(size_t)(m0 + ar) * DMODEL + k0 + 32 + ac], 0, 1);
    __syncthreads();

    bf16x16 au;
    au.h[0] = *(const v8bf*)&As[w * 16 + ln][hi * 8];
    au.h[1] = *(const v8bf*)&As[w * 16 + ln][hi * 8 + 16];
#pragma unroll
    for (int n = 0; n < 4; ++n) {
      const v16bf bfg =
          *(const v16bf*)&WT[(size_t)(n0 + n * 16 + ln) * DMODEL + k0 + hi * 16];
      acc[n] = WMMA_BF16(au.v, bfg, acc[n]);
    }
    __syncthreads();
  }

#pragma unroll
  for (int n = 0; n < 4; ++n) {
    int gn = n0 + n * 16 + ln;
    float bias = b_proj[gn];
#pragma unroll
    for (int i = 0; i < 8; ++i) {
      int gm = m0 + w * 16 + i + hi * 8;
      int b = gm >> 6, qi = gm & 63;
      if (qi < QLEN)
        out[(size_t)(b * QLEN + qi) * DMODEL + gn] = acc[n][i] + bias;
    }
  }
}

// ---------------------------------------------------------------------------
extern "C" void kernel_launch(void* const* d_in, const int* in_sizes, int n_in,
                              void* d_out, int out_size, void* d_ws, size_t ws_size,
                              hipStream_t stream) {
  (void)in_sizes; (void)n_in; (void)out_size; (void)ws_size;
  const float* x      = (const float*)d_in[0];
  const float* y      = (const float*)d_in[1];
  const float* mask   = (const float*)d_in[2];
  const float* W_attn = (const float*)d_in[3];
  const float* b_attn = (const float*)d_in[4];
  const float* W_mlp  = (const float*)d_in[5];
  const float* b_mlp  = (const float*)d_in[6];
  const float* W_proj = (const float*)d_in[7];
  const float* b_proj = (const float*)d_in[8];
  float* out = (float*)d_out;

  char* ws = (char*)d_ws;
  __bf16* WabT = (__bf16*)(ws);                                 // [2048][1024]  4 MiB
  __bf16* WmbT = (__bf16*)(ws + (size_t)4  * 1024 * 1024);      // [1024][1024]  2 MiB
  __bf16* WpbT = (__bf16*)(ws + (size_t)6  * 1024 * 1024);      // [1024][1024]  2 MiB
  __bf16* qws  = (__bf16*)(ws + (size_t)8  * 1024 * 1024);      //  2 MiB
  __bf16* aws  = (__bf16*)(ws + (size_t)10 * 1024 * 1024);      //  2 MiB
  __bf16* kws  = (__bf16*)(ws + (size_t)12 * 1024 * 1024);      // 128 MiB  [b,h,s,hd]
  __bf16* vws  = (__bf16*)(ws + (size_t)140 * 1024 * 1024);     // 128 MiB  [b,h,hd,s]

  dim3 tb(32, 8);
  transpose_cvt_kernel<<<dim3(2 * DMODEL / 32, DMODEL / 32), tb, 0, stream>>>(
      W_attn, WabT, DMODEL, 2 * DMODEL);
  transpose_cvt_kernel<<<dim3(DMODEL / 32, DMODEL / 32), tb, 0, stream>>>(
      W_mlp, WmbT, DMODEL, DMODEL);
  transpose_cvt_kernel<<<dim3(DMODEL / 32, DMODEL / 32), tb, 0, stream>>>(
      W_proj, WpbT, DMODEL, DMODEL);

  dim3 gkv((2 * DMODEL) / 128, (BATCH * SLEN) / 128);           // 16 x 512 (N fast)
  gemm_kv_kernel<<<gkv, 256, 0, stream>>>(x, WabT, b_attn, kws, vws);

  dim3 gq(DMODEL / 64, (BATCH * QPAD) / 64);                    // 16 x 16
  gemm_q_kernel<<<gq, 128, 0, stream>>>(y, WmbT, b_mlp, qws);

  attention_kernel<<<BATCH * NHEAD, 256, 0, stream>>>(kws, vws, qws, mask, aws);

  dim3 gp(DMODEL / 64, (BATCH * QPAD) / 64);                    // 16 x 16
  gemm_proj_kernel<<<gp, 128, 0, stream>>>(aws, WpbT, b_proj, out);
}
